// BiRNNEncoder_68736656605237
// MI455X (gfx1250) — matrix operand order repaired and checked by reading, not compiled, not run on hardware
//
#include <hip/hip_runtime.h>
#include <hip/hip_bf16.h>

// ---------------------------------------------------------------------------
// BiLSTM encoder for MI455X (gfx1250, wave32, WMMA).
//   V=32000 E=512 HID=1024 HD=512, B=32 U=32 T=64  ->  N = 1024 sequences.
//   * Weights converted once to bf16 in d_ws (stay L2-resident, 8.4 MB).
//   * One workgroup per (direction, 16-row batch chunk); full 64-step time
//     loop runs inside the workgroup (batch rows are independent -> no
//     global sync). h/c/max state never touches HBM.
//   * gates = bias + x_t @ W_ih^T + h @ W_hh^T via v_wmma_f32_16x16x32_bf16.
//   * Weight loads carry an opaque t-dependent stride (==0 at runtime) so the
//     compiler cannot hoist the loop-invariant B-fragments out of the time
//     loop and spill them to scratch (observed pathology in round 1).
// ---------------------------------------------------------------------------

typedef __attribute__((ext_vector_type(16))) __bf16 v16bf;
typedef __attribute__((ext_vector_type(8)))  __bf16 v8bf;
typedef __attribute__((ext_vector_type(8)))  float  v8f;

#define NSEQ   1024
#define TLEN   64
#define EDIM   512
#define HDIM   512      // hidden per direction
#define GDIM   2048     // 4*HDIM (i,f,g,o)
#define ROWS   16       // batch rows per workgroup (one WMMA M-tile)
#define STR    520      // padded LDS row stride in bf16 elems (16B-aligned rows)
#define NT     256      // threads per block (8 wave32 waves)
#define WMAT   (GDIM*EDIM)  // 1048576 elems per weight matrix

// ---- weight f32 -> bf16 conversion (runs every launch; ws is poisoned) ----
__global__ __launch_bounds__(NT) void cvt_weights_bf16(
    const float* __restrict__ w0, const float* __restrict__ w1,
    const float* __restrict__ w2, const float* __restrict__ w3,
    __bf16* __restrict__ out)
{
    int idx = blockIdx.x * NT + threadIdx.x;
    if (idx >= 4 * WMAT) return;
    int which = idx >> 20;            // WMAT == 2^20
    int off   = idx & (WMAT - 1);
    const float* src = (which == 0) ? w0 : (which == 1) ? w1
                     : (which == 2) ? w2 : w3;
    out[(size_t)which * WMAT + off] = (__bf16)src[off];
}

// A-fragment (16x32 bf16, M x K) from a row-major bf16 tile in LDS.
// Lane L: M = L%16; elements 0..15 -> K = k0 + 16*(L/16) + j (contiguous,
//         matching the B-fragment order).
__device__ inline v16bf load_a_frag(const __bf16* tile, int laneN, int half, int k0)
{
    const __bf16* p = tile + laneN * STR + k0 + 16 * half;
    v8bf lo = *(const v8bf*)(p);
    v8bf hi = *(const v8bf*)(p + 8);
    return __builtin_shufflevector(lo, hi, 0,1,2,3,4,5,6,7,8,9,10,11,12,13,14,15);
}

__device__ inline float sigmoidf_dev(float x) {
    return 1.0f / (1.0f + __expf(-x));
}

__global__ __launch_bounds__(NT) void bilstm_maxpool_kernel(
    const int*   __restrict__ tokens,   // [1024][64]  (int32 per harness)
    const float* __restrict__ emb,      // [32000][512]
    const __bf16* __restrict__ wih_f,   // [2048][512] bf16
    const __bf16* __restrict__ whh_f,   // [2048][512] bf16
    const __bf16* __restrict__ wih_b,
    const __bf16* __restrict__ whh_b,
    const float* __restrict__ bias_f,   // [2048]
    const float* __restrict__ bias_b,
    float* __restrict__ out,            // [1024][1024] (fwd 0..511 | bwd 512..1023)
    unsigned long long wstep)           // == 0 at runtime; defeats LICM over t
{
    __shared__ int tok_lds[ROWS];
    __shared__ __align__(16) __bf16 x_lds[ROWS * STR];
    __shared__ __align__(16) __bf16 h_lds[2][ROWS * STR];

    const int chunk = blockIdx.x;          // 0..63
    const int dir   = blockIdx.y;          // 0 fwd, 1 bwd
    const int n0    = chunk * ROWS;
    const int tid   = threadIdx.x;
    const int lane  = tid & 31;            // wave32
    const int wave  = tid >> 5;            // 0..7
    const int laneN = lane & 15;           // N / M-row index inside tile
    const int half  = lane >> 4;           // 0: lanes 0-15, 1: lanes 16-31

    const __bf16* wih  = dir ? wih_b  : wih_f;
    const __bf16* whh  = dir ? whh_b  : whh_f;
    const float*  bias = dir ? bias_b : bias_f;

    // zero the t=0 hidden-state buffer
    for (int i = tid; i < ROWS * STR; i += NT) h_lds[0][i] = (__bf16)0.0f;

    // bias preload: one scalar per (hd-tile q, gate g) for this lane's column
    float bias_reg[4][4];
    #pragma unroll
    for (int q = 0; q < 4; q++) {
        const int colN = (wave * 4 + q) * 16 + laneN;
        #pragma unroll
        for (int g = 0; g < 4; g++) bias_reg[q][g] = bias[g * HDIM + colN];
    }

    // per-wave persistent state: 4 HD-tiles x 8 accumulator rows
    float cstate[4][8];
    float hmax[4][8];
    #pragma unroll
    for (int q = 0; q < 4; q++)
        #pragma unroll
        for (int r = 0; r < 8; r++) { cstate[q][r] = 0.0f; hmax[q][r] = -3.0e38f; }

    #pragma unroll 1
    for (int t = 0; t < TLEN; t++) {
        __syncthreads();                       // prev step fully consumed
        const int teff = dir ? (TLEN - 1 - t) : t;
        if (tid < ROWS) tok_lds[tid] = tokens[(n0 + tid) * TLEN + teff];
        __syncthreads();

        // gather 16 embedding rows -> bf16 LDS tile (coalesced; token load
        // is a per-iteration broadcast)
        #pragma unroll 4
        for (int i = 0; i < 32; i++) {
            int flat = i * NT + tid;
            int r = flat >> 9;
            int c = flat & 511;
            x_lds[r * STR + c] = (__bf16)emb[(size_t)tok_lds[r] * EDIM + c];
        }
        __syncthreads();

        const __bf16* hread  = h_lds[t & 1];
        __bf16*       hwrite = h_lds[(t + 1) & 1];

        // t-dependent (but runtime-identical) weight bases: blocks LICM/spill
        const __bf16* wih_t = wih + (size_t)t * wstep;
        const __bf16* whh_t = whh + (size_t)t * wstep;

        #pragma unroll 1
        for (int q = 0; q < 4; q++) {
            const int hdt  = wave * 4 + q;        // HD tile 0..31
            const int colN = hdt * 16 + laneN;    // hidden column for lane N

            // C init = bias (same bias for every M row of the tile)
            v8f acc[4];
            #pragma unroll
            for (int g = 0; g < 4; g++) {
                float bv = bias_reg[q][g];
                #pragma unroll
                for (int r = 0; r < 8; r++) acc[g][r] = bv;
            }

            // gates += x_t @ W_ih^T   (K = EDIM = 512 -> 16 WMMA k-steps)
            #pragma unroll 2
            for (int kt = 0; kt < 16; kt++) {
                const int k0 = kt * 32;
                v16bf a = load_a_frag(x_lds, laneN, half, k0);
                #pragma unroll
                for (int g = 0; g < 4; g++) {
                    // B frag: lane holds column N=colN; 16 contiguous bf16 of
                    // the row-major weight row (K = k0 + 16*half + 0..15).
                    const __bf16* bp = wih_t + (size_t)(g * HDIM + colN) * EDIM
                                             + k0 + 16 * half;
                    v16bf b = *(const v16bf*)bp;
                    acc[g] = __builtin_amdgcn_wmma_f32_16x16x32_bf16(
                        false, a, false, b, (short)0, acc[g], false, false);
                }
            }
            // gates += h_{t-1} @ W_hh^T  (K = HDIM = 512)
            #pragma unroll 2
            for (int kt = 0; kt < 16; kt++) {
                const int k0 = kt * 32;
                v16bf a = load_a_frag(hread, laneN, half, k0);
                #pragma unroll
                for (int g = 0; g < 4; g++) {
                    const __bf16* bp = whh_t + (size_t)(g * HDIM + colN) * HDIM
                                             + k0 + 16 * half;
                    v16bf b = *(const v16bf*)bp;
                    acc[g] = __builtin_amdgcn_wmma_f32_16x16x32_bf16(
                        false, a, false, b, (short)0, acc[g], false, false);
                }
            }

            // LSTM cell (PyTorch gate order i,f,g,o), running max over t.
            // C/D layout: VGPR r -> row M = r + 8*half, col N = laneN.
            #pragma unroll
            for (int r = 0; r < 8; r++) {
                float iv = sigmoidf_dev(acc[0][r]);
                float fv = sigmoidf_dev(acc[1][r]);
                float gv = tanhf(acc[2][r]);
                float ov = sigmoidf_dev(acc[3][r]);
                float cn = fv * cstate[q][r] + iv * gv;
                cstate[q][r] = cn;
                float hn = ov * tanhf(cn);
                hmax[q][r] = fmaxf(hmax[q][r], hn);
                hwrite[(r + 8 * half) * STR + colN] = (__bf16)hn;
            }
        }
    }

    // write max-pooled hidden states: out[n][dir*512 + hd]
    #pragma unroll
    for (int q = 0; q < 4; q++) {
        const int colN = (wave * 4 + q) * 16 + laneN;
        #pragma unroll
        for (int r = 0; r < 8; r++) {
            const int m = r + 8 * half;
            out[(size_t)(n0 + m) * 1024 + dir * HDIM + colN] = hmax[q][r];
        }
    }
}

extern "C" void kernel_launch(void* const* d_in, const int* in_sizes, int n_in,
                              void* d_out, int out_size, void* d_ws, size_t ws_size,
                              hipStream_t stream)
{
    const int*   tokens = (const int*)  d_in[0];   // input_w  [32][32][64]
    const float* emb    = (const float*)d_in[1];   // [32000][512]
    const float* w_ih_f = (const float*)d_in[2];   // [2048][512]
    const float* w_hh_f = (const float*)d_in[3];   // [2048][512]
    const float* b_f    = (const float*)d_in[4];   // [2048]
    const float* w_ih_b = (const float*)d_in[5];
    const float* w_hh_b = (const float*)d_in[6];
    const float* b_b    = (const float*)d_in[7];
    float* out = (float*)d_out;                    // [1024][1024]

    __bf16* wbf = (__bf16*)d_ws;                   // 4 x 2 MB bf16 weights

    const int total = 4 * WMAT;
    cvt_weights_bf16<<<(total + NT - 1) / NT, NT, 0, stream>>>(
        w_ih_f, w_hh_f, w_ih_b, w_hh_b, wbf);

    bilstm_maxpool_kernel<<<dim3(NSEQ / ROWS, 2), NT, 0, stream>>>(
        tokens, emb,
        wbf + 0 * (size_t)WMAT, wbf + 1 * (size_t)WMAT,
        wbf + 2 * (size_t)WMAT, wbf + 3 * (size_t)WMAT,
        b_f, b_b, out, 0ull);

    (void)in_sizes; (void)n_in; (void)out_size; (void)ws_size;
}
